// PWSim_43748536877181
// MI455X (gfx1250) — matrix operand order repaired and checked by reading, not compiled, not run on hardware
//
#include <hip/hip_runtime.h>
#include <stdint.h>

#define NEL   14
#define CHN   18
#define BATCH 4
#define HDIM  512
#define WDIM  512
#define HWSZ  (HDIM*WDIM)

enum { E_EMPTY=0, E_WALL, E_SAND, E_WATER, E_GAS, E_WOOD, E_ICE, E_FIRE,
       E_PLANT, E_STONE, E_LAVA, E_ACID, E_DUST, E_CLONER };

__constant__ float         c_dens[NEL] = {0.f,5.f,3.f,2.f,-1.f,5.f,5.f,0.f,5.f,4.f,3.f,2.f,3.f,5.f};
__constant__ unsigned char c_grav[NEL] = {1,0,1,1,1,0,1,1,0,1,1,1,1,0};

__device__ __forceinline__ bool is_blockE(int id){ return !(id==E_EMPTY||id==E_WALL||id==E_ACID||id==E_CLONER); }
__device__ __forceinline__ bool is_burnE (int id){ return id==E_WOOD||id==E_PLANT||id==E_GAS||id==E_DUST; }
__device__ __forceinline__ bool is_flE   (int id){ return id==E_FIRE||id==E_LAVA; }

// ---- CDNA5 async global->LDS copy (ASYNCcnt path) -------------------------
__device__ __forceinline__ void async_g2l_b32(void* ldsDst, const void* gSrc){
  uint32_t l = (uint32_t)(uintptr_t)ldsDst;   // low 32 bits of generic ptr = LDS offset
  uint64_t g = (uint64_t)(uintptr_t)gSrc;
  asm volatile("global_load_async_to_lds_b32 %0, %1, off" :: "v"(l), "v"(g) : "memory");
}
__device__ __forceinline__ void wait_async0(){
  asm volatile("s_wait_asynccnt 0x0" ::: "memory");
}

// ---- CDNA5 Tensor Data Mover: stage a small 2D f32 tensor into LDS --------
typedef unsigned int u32x4 __attribute__((ext_vector_type(4)));
typedef int          i32x4 __attribute__((ext_vector_type(4)));
typedef int          i32x8 __attribute__((ext_vector_type(8)));

__device__ __forceinline__ void tdm_load_2d_f32(float* ldsDst, const float* gSrc,
                                                int dim0, int dim1)
{
#if __has_builtin(__builtin_amdgcn_tensor_load_to_lds)
  uint32_t lds_off = (uint32_t)(uintptr_t)ldsDst;
  uint64_t ga      = (uint64_t)(uintptr_t)gSrc;
  u32x4 g0;
  g0.x = 1u;                                           // count=1, user mode
  g0.y = lds_off;                                      // lds_addr [63:32]
  g0.z = (uint32_t)(ga & 0xffffffffu);                 // global_addr [95:64]
  g0.w = (uint32_t)((ga >> 32) & 0x01ffffffu)          // global_addr [120:96]
       | (2u << 30);                                   // type=2 ("image")
  i32x8 g1;
  g1[0] = (2 << 16);                                   // wg_mask=0, data_size=4B
  g1[1] = (dim0 & 0xffff) << 16;                       // tensor_dim0 low16 @bit48
  g1[2] = ((dim0 >> 16) & 0xffff) | ((dim1 & 0xffff) << 16); // dim0 hi / dim1 lo
  g1[3] = ((dim1 >> 16) & 0xffff) | (dim0 << 16);      // dim1 hi / tile_dim0
  g1[4] = dim1 & 0xffff;                               // tile_dim1 (tile_dim2=0)
  g1[5] = dim0;                                        // tensor_dim0_stride lo32
  g1[6] = 0;                                           // stride hi / dim1_stride lo
  g1[7] = 0;
  i32x4 z4 = {0,0,0,0};
#if __clang_major__ >= 23
  i32x8 z8 = {0,0,0,0,0,0,0,0};
  __builtin_amdgcn_tensor_load_to_lds(g0, g1, z4, z4, z8, 0);
#else
  __builtin_amdgcn_tensor_load_to_lds(g0, g1, z4, z4, 0);
#endif
  __builtin_amdgcn_s_wait_tensorcnt(0);
#else
  for (int i = (int)(threadIdx.x & 31); i < dim0*dim1; i += 32)
    ldsDst[i] = gSrc[i];
#endif
}

// ===========================================================================
// K1: compact 18-channel world -> {id:u8, ca:u8, mom:f32} (4 cells/thread)
// ===========================================================================
__global__ __launch_bounds__(256) void k_compact(const float4* __restrict__ world4,
                                                 uchar4* __restrict__ ids4,
                                                 uchar4* __restrict__ ca4,
                                                 float4* __restrict__ mom4)
{
  const int QP = HWSZ/4;                 // float4 quads per plane
  int q  = blockIdx.x*256 + threadIdx.x; // quad index over all batches
  int b  = q / QP;
  int hq = q - b*QP;
  const float4* wb = world4 + (size_t)b*CHN*QP;
  int i0=0, i1=0, i2=0, i3=0;
  #pragma unroll
  for (int c = 0; c < NEL; ++c){
    float4 v = wb[(size_t)c*QP + hq];
    if (v.x > 0.5f) i0 = c;
    if (v.y > 0.5f) i1 = c;
    if (v.z > 0.5f) i2 = c;
    if (v.w > 0.5f) i3 = c;
  }
  ids4[q] = make_uchar4((unsigned char)i0,(unsigned char)i1,(unsigned char)i2,(unsigned char)i3);
  mom4[q] = wb[(size_t)16*QP + hq];
  float4 cv = wb[(size_t)17*QP + hq];
  ca4[q] = make_uchar4((unsigned char)(int)(cv.x+0.5f),(unsigned char)(int)(cv.y+0.5f),
                       (unsigned char)(int)(cv.z+0.5f),(unsigned char)(int)(cv.w+0.5f));
}

// ===========================================================================
// K2: acid + fire (radius-4 fused stencil, 16x16 tile, halo 4)
//  - roll() neighbors wrap; conv3x3 terms are zero-padded at image borders
// ===========================================================================
#define HALO2 4
#define S2 (16 + 2*HALO2)

__global__ __launch_bounds__(256) void k_acid_fire(
    const unsigned char* __restrict__ idsIn, const unsigned char* __restrict__ caIn,
    float* __restrict__ mom,
    const float* __restrict__ rint, const float* __restrict__ relem,
    const float* __restrict__ velIn,
    unsigned char* __restrict__ idsOut, unsigned char* __restrict__ caOut,
    float* __restrict__ velOut)
{
  __shared__ unsigned char sid0[S2*S2];
  __shared__ float         srnd[S2*S2];
  __shared__ unsigned char sid1[S2*S2];
  __shared__ unsigned char sid2[S2*S2];
  __shared__ unsigned char sbfn[S2*S2];
  __shared__ unsigned char sdfn[S2*S2];

  int b  = blockIdx.z;
  int h0 = blockIdx.y*16, w0 = blockIdx.x*16;
  int tid = threadIdx.y*16 + threadIdx.x;
  size_t base = (size_t)b*HWSZ;

  for (int i = tid; i < S2*S2; i += 256){
    int y = i/S2, x = i - y*S2;
    int gh = (h0+y-HALO2)&(HDIM-1), gw = (w0+x-HALO2)&(WDIM-1);
    size_t g = base + (size_t)gh*WDIM + gw;
    sid0[i] = idsIn[g];
    srnd[i] = rint[g];
  }
  __syncthreads();

  // ---- acid (valid region [1..22]^2), rolls wrap -------------------------
  for (int i = tid; i < S2*S2; i += 256){
    int y = i/S2, x = i - y*S2;
    if (y < 1 || y > 22 || x < 1 || x > 22){ sid1[i] = sid0[i]; continue; }
    int id = sid0[i]; float r = srnd[i];
    int up = sid0[i-S2], dn = sid0[i+S2];
    bool da = (id==E_ACID) && (r < 0.2f) && (is_blockE(dn) || is_blockE(up));
    bool db = is_blockE(id) && ((srnd[i-S2] < 0.2f && up==E_ACID) ||
                                (srnd[i+S2] < 0.2f && dn==E_ACID));
    sid1[i] = (da || db) ? (unsigned char)E_EMPTY : (unsigned char)id;
  }
  __syncthreads();

  // ---- fire stage 1 (valid region [2..21]^2) -----------------------------
  for (int i = tid; i < S2*S2; i += 256){
    int y = i/S2, x = i - y*S2;
    if (y < 2 || y > 21 || x < 2 || x > 21){ sid2[i]=sid1[i]; sbfn[i]=0; sdfn[i]=0; continue; }
    int hh = (h0+y-HALO2)&(HDIM-1), ww = (w0+x-HALO2)&(WDIM-1);
    int hf = 0;
    #pragma unroll
    for (int dy = -1; dy <= 1; ++dy)
      #pragma unroll
      for (int dx = -1; dx <= 1; ++dx){
        if ((unsigned)(hh+dy) >= HDIM || (unsigned)(ww+dx) >= WDIM) continue; // zero pad
        hf += is_flE(sid1[i + dy*S2 + dx]) ? 1 : 0;
      }
    bool hasfire = hf > 0;
    int id1 = sid1[i]; float r = srnd[i];
    bool dbw = (id1==E_WOOD ) && (r < 0.05f);
    bool dbp = (id1==E_PLANT) && (r < 0.2f);
    bool dbg = (id1==E_GAS  ) && (r < 0.2f);
    bool dbd = (id1==E_DUST);
    bool dice= (id1==E_ICE  ) && (r < 0.2f) && hasfire;
    bool burn= (dbw||dbp||dbg||dbd) && hasfire;
    sid2[i] = burn ? (unsigned char)E_FIRE : (dice ? (unsigned char)E_WATER : (unsigned char)id1);
    sbfn[i] = burn ? 1 : 0;
    sdfn[i] = (dbd && hasfire) ? 1 : 0;
  }
  __syncthreads();

  // ---- fire stage 2 + outputs (center cell per thread) -------------------
  {
    int y = threadIdx.y + HALO2, x = threadIdx.x + HALO2;
    int i = y*S2 + x;
    int gh = h0 + threadIdx.y, gw = w0 + threadIdx.x;
    size_t g = base + (size_t)gh*WDIM + gw;

    // in_fire_range = conv3x3( conv3x3(burnables)*fire_and_lava + lava )
    float ifr = 0.f;
    #pragma unroll
    for (int dy = -1; dy <= 1; ++dy)
      #pragma unroll
      for (int dx = -1; dx <= 1; ++dx){
        if ((unsigned)(gh+dy) >= HDIM || (unsigned)(gw+dx) >= WDIM) continue;
        int j = i + dy*S2 + dx;
        int bc = 0;
        #pragma unroll
        for (int my = -1; my <= 1; ++my)
          #pragma unroll
          for (int mx = -1; mx <= 1; ++mx){
            if ((unsigned)(gh+dy+my) >= HDIM || (unsigned)(gw+dx+mx) >= WDIM) continue;
            bc += is_burnE(sid2[j + my*S2 + mx]) ? 1 : 0;
          }
        float fl = is_flE(sid1[j]) ? 1.f : 0.f;
        ifr += (float)bc*fl + (sid2[j]==E_LAVA ? 1.f : 0.f);
      }

    int id2 = sid2[i]; float r = srnd[i];
    int id3 = (id2==E_EMPTY && ifr > 0.f && r < 0.3f) ? E_FIRE : id2;

    int hb = 0;
    #pragma unroll
    for (int dy = -1; dy <= 1; ++dy)
      #pragma unroll
      for (int dx = -1; dx <= 1; ++dx){
        if ((unsigned)(gh+dy) >= HDIM || (unsigned)(gw+dx) >= WDIM) continue;
        hb += is_burnE(sid2[i + dy*S2 + dx]) ? 1 : 0;
      }
    float re = relem[g];
    int id4 = (id3==E_FIRE && re < 0.4f && hb == 0) ? E_EMPTY : id3;

    bool replaced = (sid1[i]!=sid0[i]) || (sid2[i]!=sid1[i]) || (id3!=id2) || (id4!=id3);
    idsOut[g] = (unsigned char)id4;
    caOut[g]  = replaced ? 0 : caIn[g];
    if (replaced) mom[g] = 0.f;

    // velocity updates (rolls wrap; sbfn/sdfn halo cells wrap correctly)
    float dvy = 2.f*((float)sbfn[i-S2]-(float)sbfn[i+S2]) + 20.f*((float)sdfn[i-S2]-(float)sdfn[i+S2]);
    float dvx = 2.f*((float)sbfn[i-1 ]-(float)sbfn[i+1 ]) + 20.f*((float)sdfn[i-1 ]-(float)sdfn[i+1 ]);
    size_t vb = (size_t)b*2*HWSZ + (size_t)gh*WDIM + gw;
    velOut[vb]        = velIn[vb]        + dvy;
    velOut[vb + HWSZ] = velIn[vb + HWSZ] + dvx;
  }
}

// ===========================================================================
// K3: fluid flow — one full row per workgroup, resident in LDS.
//     10 sequential swap steps; f32 rows staged via async global->LDS DMA.
// ===========================================================================
__global__ __launch_bounds__(512) void k_fluid(
    unsigned char* __restrict__ ids, unsigned char* __restrict__ ca,
    float* __restrict__ mom,
    const float* __restrict__ rmov, const float* __restrict__ dgrav)
{
  __shared__ unsigned char sid[WDIM];
  __shared__ unsigned char sca[WDIM];
  __shared__ unsigned char sng[WDIM];   // not-did-gravity
  __shared__ float         smom[WDIM];
  __shared__ float         snew[WDIM];
  __shared__ float         srm[WDIM];

  int row = blockIdx.x;
  int b = row / HDIM, h = row - b*HDIM;
  int w = threadIdx.x;
  size_t g = (size_t)b*HWSZ + (size_t)h*WDIM + w;

  // async DMA for the two f32 rows (ASYNCcnt), scalar path for the u8 rows
  async_g2l_b32(&smom[w], mom  + g);
  async_g2l_b32(&srm [w], rmov + g);
  sid[w] = ids[g];
  sca[w] = ca[g];
  sng[w] = (dgrav[g] <= 0.f) ? 1 : 0;
  snew[w] = 0.f;
  wait_async0();
  __syncthreads();

  const int elems[5] = {E_EMPTY, E_WATER, E_GAS, E_LAVA, E_ACID};
  for (int ei = 0; ei < 5; ++ei){
    #pragma unroll
    for (int fd = 0; fd < 2; ++fd){
      const bool fl = (fd == 0);              // fall_left
      const int  d  = fl ? -1 : 1;            // gdir offset
      const int  e  = elems[ei];
      int wp = (w + d) & (WDIM-1);            // gdir neighbor
      int wm = (w - d) & (WDIM-1);            // gnot neighbor

      bool fdir_w  = (srm[w ] + smom[w ] + snew[w ]) > 0.5f;
      bool fdir_wm = (srm[wm] + smom[wm] + snew[wm]) > 0.5f;
      bool match_w  = fl ? fdir_w  : !fdir_w;
      bool match_wm = fl ? fdir_wm : !fdir_wm;

      float dw = c_dens[sid[w]], dp = c_dens[sid[wp]], dm = c_dens[sid[wm]];
      bool gvw = c_grav[sid[w]] != 0, gvp = c_grav[sid[wp]] != 0, gvm = c_grav[sid[wm]] != 0;

      bool dbl = match_w  && (sid[w ]==e) && sng[w ] && (dw - dp > 0.f) && gvp && gvw;
      bool dbr = match_wm && (sid[wm]==e) && sng[wm] && (dm - dw > 0.f) && gvm && gvw;

      unsigned char nid = dbl ? sid[wp] : (dbr ? sid[wm] : sid[w]);
      unsigned char nca = dbl ? sca[wp] : (dbr ? sca[wm] : sca[w]);
      float        nmom = dbl ? smom[wp]: (dbr ? smom[wm]: smom[w]);
      float add = dbr ? (fl ? 2.f : -2.f) : 0.f;

      __syncthreads();
      sid[w] = nid; sca[w] = nca; smom[w] = nmom; snew[w] += add;
      __syncthreads();
    }
  }
  ids[g] = sid[w];
  ca [g] = sca[w];
  mom[g] = snew[w];      // world mom channel := new_mom
}

// ===========================================================================
// K4: cloner — 16x16 tile, halo 2
// ===========================================================================
#define HALO4 2
#define S4 (16 + 2*HALO4)

__global__ __launch_bounds__(256) void k_cloner(
    const unsigned char* __restrict__ idsIn, const unsigned char* __restrict__ caIn,
    float* __restrict__ mom,
    unsigned char* __restrict__ idsOut, unsigned char* __restrict__ caOut)
{
  __shared__ unsigned char sid[S4*S4];
  __shared__ unsigned char sac[S4*S4];
  __shared__ unsigned char sas[S4*S4];

  int b  = blockIdx.z;
  int h0 = blockIdx.y*16, w0 = blockIdx.x*16;
  int tid = threadIdx.y*16 + threadIdx.x;
  size_t base = (size_t)b*HWSZ;

  for (int i = tid; i < S4*S4; i += 256){
    int y = i/S4, x = i - y*S4;
    int gh = (h0+y-HALO4)&(HDIM-1), gw = (w0+x-HALO4)&(WDIM-1);
    size_t g = base + (size_t)gh*WDIM + gw;
    sid[i] = idsIn[g];
    sac[i] = caIn[g];
  }
  __syncthreads();

  // assignment pass (valid [1..18]^2); directions: below, above, left, right
  for (int i = tid; i < S4*S4; i += 256){
    int y = i/S4, x = i - y*S4;
    if (y < 1 || y > 18 || x < 1 || x > 18){ sas[i] = 0; continue; }
    int id = sid[i];
    unsigned char a = 0;
    unsigned char as = 0;
    if (id == E_CLONER){
      a = sac[i];
      const int offs[4] = {+S4, -S4, -1, +1};
      #pragma unroll
      for (int k = 0; k < 4; ++k)
        if (a == 0 || a == 13) a = sid[i + offs[k]];
      as = (a != 0 && a != 13) ? 1 : 0;
    }
    sac[i] = a;   // non-cloner cells: new_ca = 0
    sas[i] = as;
  }
  __syncthreads();

  {
    int y = threadIdx.y + HALO4, x = threadIdx.x + HALO4;
    int i = y*S4 + x;
    int gh = h0 + threadIdx.y, gw = w0 + threadIdx.x;
    size_t g = base + (size_t)gh*WDIM + gw;
    int id = sid[i];
    bool cloned = false;
    if (id == E_EMPTY){
      const int offs[4] = {+S4, -S4, -1, +1};
      #pragma unroll
      for (int k = 0; k < 4; ++k){
        int j = i + offs[k];
        if (!cloned && sas[j]){ id = sac[j]; cloned = true; }
      }
    }
    idsOut[g] = (unsigned char)id;
    caOut[g]  = sac[i];
    if (cloned) mom[g] = 0.f;
  }
}

// ===========================================================================
// K5: expand {id,ca,mom} -> 18-channel world.
//     Channels 0..15 = one_hot(id)[16x14] @ elem_table[14x16]: exact f16 WMMA.
//     elem_table staged into LDS by the Tensor Data Mover (TENSORcnt path).
// ===========================================================================
typedef __attribute__((ext_vector_type(16))) _Float16 v16h;
typedef __attribute__((ext_vector_type(8)))  float    v8f;

__global__ __launch_bounds__(256) void k_expand(
    const unsigned char* __restrict__ ids, const unsigned char* __restrict__ ca,
    const float* __restrict__ mom, const float* __restrict__ table,
    float* __restrict__ outw)
{
  __shared__ float sTab[NEL*CHN];     // 14 x 18 element table
  __shared__ float sOut[8][16][16];   // [wave(seg)][channel][cell]

  int b = blockIdx.z, h = blockIdx.y, w0 = blockIdx.x*128;
  int tid = threadIdx.x;
  int wave = tid >> 5, lane = tid & 31;
  size_t rowbase = (size_t)b*HWSZ + (size_t)h*WDIM;

  if (tid < 32)                        // wave 0 drives the TDM DMA
    tdm_load_2d_f32(sTab, table, CHN, NEL);
  __syncthreads();

  // A: 16x32 one-hot (K = element id; K>=14 zero padded) — branchless, all VGPR
  union { v16h v; _Float16 e[16]; } A, B;
  int m  = lane & 15;
  int id = ids[rowbase + w0 + wave*16 + m];
  int kl = (lane < 16) ? id : (id - 16);
  #pragma unroll
  for (int k = 0; k < 16; ++k)
    A.e[k] = (kl == k) ? (_Float16)1.0f : (_Float16)0.0f;

  // B: 32x16 elem_table (rows K=0..13 real; rest zero). lane holds col n=lane&15,
  // K = e + 16*(lane/16). Branchless select on LDS-staged table.
  int n = lane & 15;
  bool lo = (lane < 16);
  #pragma unroll
  for (int k = 0; k < NEL; ++k)
    B.e[k] = lo ? (_Float16)sTab[k*CHN + n] : (_Float16)0.0f;
  #pragma unroll
  for (int k = NEL; k < 16; ++k)
    B.e[k] = (_Float16)0.0f;

  v8f C = {0.f,0.f,0.f,0.f,0.f,0.f,0.f,0.f};
  union { v8f v; float e[8]; } D;
  D.v = __builtin_amdgcn_wmma_f32_16x16x32_f16(false, A.v, false, B.v,
                                               (short)0, C, false, false);

  // D layout: lane -> col N=lane&15, vgpr v -> row M = v + 8*(lane/16)
  int mb = lo ? 0 : 8;
  #pragma unroll
  for (int v = 0; v < 8; ++v)
    sOut[wave][n][mb + v] = D.e[v];
  __syncthreads();

  // coalesced stores of channels 0..15
  const size_t plane = (size_t)HWSZ;
  float* ob = outw + (size_t)b*CHN*plane + (size_t)h*WDIM + w0;
  for (int idx = tid; idx < 16*128; idx += 256){
    int c = idx >> 7, j = idx & 127;
    ob[(size_t)c*plane + j] = sOut[j >> 4][c][j & 15];
  }
  // channels 16 (momentum) and 17 (cloner assignment)
  for (int idx = tid; idx < 128; idx += 256){
    size_t g = rowbase + w0 + idx;
    ob[(size_t)16*plane + idx] = mom[g];
    ob[(size_t)17*plane + idx] = (float)ca[g];
  }
}

// ===========================================================================
extern "C" void kernel_launch(void* const* d_in, const int* in_sizes, int n_in,
                              void* d_out, int out_size, void* d_ws, size_t ws_size,
                              hipStream_t stream)
{
  (void)in_sizes; (void)n_in; (void)out_size; (void)ws_size;
  const float* world = (const float*)d_in[0];
  const float* rmov  = (const float*)d_in[1];
  const float* rint  = (const float*)d_in[2];
  const float* relem = (const float*)d_in[3];
  const float* vel   = (const float*)d_in[4];
  const float* dgrav = (const float*)d_in[5];
  const float* table = (const float*)d_in[6];

  float* outw = (float*)d_out;
  float* outv = outw + (size_t)BATCH*CHN*HWSZ;

  // workspace layout (8 MB total)
  unsigned char* ids0 = (unsigned char*)d_ws;
  unsigned char* ca0  = ids0 + (1u << 20);
  unsigned char* ids1 = ids0 + (2u << 20);
  unsigned char* ca1  = ids0 + (3u << 20);
  float*         mom  = (float*)(ids0 + (4u << 20));

  const int ncell = BATCH*HWSZ;
  dim3 tileGrid(WDIM/16, HDIM/16, BATCH);

  k_compact  <<<(ncell/4)/256, 256, 0, stream>>>((const float4*)world, (uchar4*)ids0,
                                                 (uchar4*)ca0, (float4*)mom);
  k_acid_fire<<<tileGrid, dim3(16,16), 0, stream>>>(ids0, ca0, mom, rint, relem, vel,
                                                    ids1, ca1, outv);
  k_fluid    <<<BATCH*HDIM, 512, 0, stream>>>(ids1, ca1, mom, rmov, dgrav);
  k_cloner   <<<tileGrid, dim3(16,16), 0, stream>>>(ids1, ca1, mom, ids0, ca0);
  k_expand   <<<dim3(WDIM/128, HDIM, BATCH), 256, 0, stream>>>(ids0, ca0, mom, table, outw);
}
